// GINEncoder_42666205118858
// MI455X (gfx1250) — matrix operand order repaired and checked by reading, not compiled, not run on hardware
//
#include <hip/hip_runtime.h>
#include <hip/hip_bf16.h>

#define HD 128   // hidden dim
#define LNUM 3

typedef __attribute__((ext_vector_type(16))) __bf16 v16bf;
typedef __attribute__((ext_vector_type(8)))  __bf16 v8bf;
typedef __attribute__((ext_vector_type(4)))  __bf16 v4bf;
typedef __attribute__((ext_vector_type(8)))  float  v8f;

// ---------------------------------------------------------------------------
// A 16x32 bf16 fragment (contiguous k order chosen here; B is packed to match):
// lane m (= lane&15) holds row m; K = k0 + 16*hi + j (j<16).
// Both 8-element groups are contiguous & 16B aligned -> two b128 LDS loads.
__device__ __forceinline__ v16bf load_a_frag(const __bf16* sA, int lane, int k0) {
  const int row = lane & 15;
  const int hi8 = (lane >> 4) * 16;
  const __bf16* p = sA + row * HD + k0 + hi8;
  const v8bf a0 = *(const v8bf*)(p);
  const v8bf a1 = *(const v8bf*)(p + 8);
  v16bf a;
#pragma unroll
  for (int j = 0; j < 8; ++j) { a[j] = a0[j]; a[8 + j] = a1[j]; }
  return a;
}

// B fragments are pre-packed fragment-major: lane's 16 bf16 contiguous (32B).
__device__ __forceinline__ v16bf load_b_packed(const __bf16* sWp, int lane,
                                               int kk, int nt) {
  return *(const v16bf*)(sWp + ((((nt << 2) + kk) << 5) + lane) * 16);
}

// Convert W (f32, row-major [K=128][N=128]) into packed B-fragment layout:
// element (k, n) -> fragment (kk = k/32, nt = n/16), lane = (k%32)/16*16 + n%16,
// j = k%16.  (B 32x16: col = lane&15, K = k0 + 16*(lane>>4) + j)
__device__ __forceinline__ void stage_w_packed(const float* __restrict__ W,
                                               __bf16* __restrict__ sWp, int tid) {
  for (int idx = tid; idx < HD * HD; idx += 256) {
    const int k = idx >> 7, n = idx & 127;
    const int nt = n >> 4, lo = n & 15;
    const int kk = k >> 5, kr = k & 31;
    const int lane = ((kr >> 4) << 4) | lo;
    const int j = kr & 15;
    sWp[((((nt << 2) + kk) << 5) + lane) * 16 + j] = (__bf16)W[idx];
  }
}

__device__ __forceinline__ void atomAddF(float* p, float v) {
  unsafeAtomicAdd(p, v);   // native global_atomic_add_f32 path
}

// ---------------------------------------------------------------------------
// h = x @ in_W + in_b       (one wave per node, lane handles 4 columns)
__global__ __launch_bounds__(256) void inproj_kernel(
    const float* __restrict__ x, const float* __restrict__ in_W,
    const float* __restrict__ in_b, float* __restrict__ h, int N) {
  const int t = blockIdx.x * blockDim.x + threadIdx.x;
  const int node = t >> 5;
  if (node >= N) return;
  const int c = (t & 31) * 4;
  float4 acc = *(const float4*)(in_b + c);
#pragma unroll
  for (int k = 0; k < 16; ++k) {
    const float xv = x[node * 16 + k];
    const float4 w = *(const float4*)(in_W + k * HD + c);
    acc.x += xv * w.x; acc.y += xv * w.y; acc.z += xv * w.z; acc.w += xv * w.w;
  }
  *(float4*)(h + (size_t)node * HD + c) = acc;
}

__global__ __launch_bounds__(256) void zero4_kernel(float4* __restrict__ p, int n4) {
  const int t = blockIdx.x * blockDim.x + threadIdx.x;
  if (t < n4) p[t] = make_float4(0.f, 0.f, 0.f, 0.f);
}

// ---------------------------------------------------------------------------
// Fused GINE message: msg = relu(h[src] + edge_attr@edge_W + edge_b);
// agg[dst] += msg.  One wave per edge, lane handles 4 columns (float4).
__global__ __launch_bounds__(256) void msg_kernel(
    const float* __restrict__ h, const float* __restrict__ edge_attr,
    const int* __restrict__ edge_index, const float* __restrict__ edge_W,
    const float* __restrict__ edge_b, float* __restrict__ agg, int E) {
  const unsigned t = blockIdx.x * blockDim.x + threadIdx.x;
  const int e = (int)(t >> 5);
  if (e >= E) return;
  const int c = (int)(t & 31) * 4;
  const int s = edge_index[e];
  const int d = edge_index[E + e];
  const float a0 = edge_attr[e * 2 + 0];
  const float a1 = edge_attr[e * 2 + 1];
  const float4 w0 = *(const float4*)(edge_W + 0 * HD + c);
  const float4 w1 = *(const float4*)(edge_W + 1 * HD + c);
  const float4 eb = *(const float4*)(edge_b + c);
  const float4 hv = *(const float4*)(h + (size_t)s * HD + c);
  const float m0 = fmaxf(hv.x + a0 * w0.x + a1 * w1.x + eb.x, 0.0f);
  const float m1 = fmaxf(hv.y + a0 * w0.y + a1 * w1.y + eb.y, 0.0f);
  const float m2 = fmaxf(hv.z + a0 * w0.z + a1 * w1.z + eb.z, 0.0f);
  const float m3 = fmaxf(hv.w + a0 * w0.w + a1 * w1.w + eb.w, 0.0f);
  float* ap = agg + (size_t)d * HD + c;
  atomAddF(ap + 0, m0); atomAddF(ap + 1, m1);
  atomAddF(ap + 2, m2); atomAddF(ap + 3, m3);
}

// ---------------------------------------------------------------------------
// Fused node update:  z = (1+eps)*h + agg
//   t  = relu(z @ W1 + b1);  z2 = t @ W2 + b2;  z2 = BN(z2)
//   h  = relu(z2) + h
// 256 threads = 8 waves; each wave owns a 16-row tile; WMMA bf16 16x16x32.
__global__ __launch_bounds__(256) void gine_mlp_kernel(
    float* __restrict__ h, const float* __restrict__ agg,
    const float* __restrict__ W1, const float* __restrict__ b1,
    const float* __restrict__ W2, const float* __restrict__ b2,
    const float* __restrict__ gamma, const float* __restrict__ beta,
    const float* __restrict__ mean, const float* __restrict__ var,
    const float* __restrict__ eps_arr, int layer, int nTiles) {
  __shared__ __bf16 sWp[HD * HD];       // 32 KB packed fragments, W1 then W2
  __shared__ __bf16 sA[8][16 * HD];     // 32 KB, per-wave A staging

  const int tid  = threadIdx.x;
  const int wave = tid >> 5;
  const int lane = tid & 31;
  const int tile = blockIdx.x * 8 + wave;
  const bool valid = (tile < nTiles);
  const int r0 = tile * 16;
  __bf16* aw = &sA[wave][0];
  const int hi = lane >> 4;
  const int lo = lane & 15;

  // ---- stage W1 into packed-fragment LDS layout ----
  stage_w_packed(W1, sWp, tid);
  __syncthreads();

  v8f c1[8];
  if (valid) {
    const float oneeps = 1.0f + eps_arr[layer];
    // stage A = (1+eps)*h + agg  (bf16, row-major 16x128)
    const int c = lane * 4;
#pragma unroll
    for (int r = 0; r < 16; ++r) {
      const float4 hv = *(const float4*)(h   + (size_t)(r0 + r) * HD + c);
      const float4 av = *(const float4*)(agg + (size_t)(r0 + r) * HD + c);
      v4bf zv;
      zv[0] = (__bf16)(oneeps * hv.x + av.x);
      zv[1] = (__bf16)(oneeps * hv.y + av.y);
      zv[2] = (__bf16)(oneeps * hv.z + av.z);
      zv[3] = (__bf16)(oneeps * hv.w + av.w);
      *(v4bf*)(aw + r * HD + c) = zv;
    }
    // GEMM1: c1 = A @ W1
    v16bf af[4];
#pragma unroll
    for (int kk = 0; kk < 4; ++kk) af[kk] = load_a_frag(aw, lane, kk * 32);
#pragma unroll
    for (int nt = 0; nt < 8; ++nt) {
      v8f acc = {};
#pragma unroll
      for (int kk = 0; kk < 4; ++kk) {
        const v16bf bf_ = load_b_packed(sWp, lane, kk, nt);
        acc = __builtin_amdgcn_wmma_f32_16x16x32_bf16(
            false, af[kk], false, bf_, (short)0, acc, false, false);
      }
      c1[nt] = acc;
    }
    // A2 = relu(c1 + b1) back into per-wave staging (C/D layout scatter)
#pragma unroll
    for (int nt = 0; nt < 8; ++nt) {
#pragma unroll
      for (int v = 0; v < 8; ++v) {
        const int row = v + 8 * hi;
        const int col = nt * 16 + lo;
        const float tv = c1[nt][v] + b1[col];
        aw[row * HD + col] = (__bf16)fmaxf(tv, 0.0f);
      }
    }
  }
  __syncthreads();
  // ---- stage W2 into packed-fragment LDS layout ----
  stage_w_packed(W2, sWp, tid);
  __syncthreads();

  if (valid) {
    // GEMM2: c2 = A2 @ W2
    v16bf af[4];
#pragma unroll
    for (int kk = 0; kk < 4; ++kk) af[kk] = load_a_frag(aw, lane, kk * 32);
#pragma unroll
    for (int nt = 0; nt < 8; ++nt) {
      v8f acc = {};
#pragma unroll
      for (int kk = 0; kk < 4; ++kk) {
        const v16bf bf_ = load_b_packed(sWp, lane, kk, nt);
        acc = __builtin_amdgcn_wmma_f32_16x16x32_bf16(
            false, af[kk], false, bf_, (short)0, acc, false, false);
      }
      // fused epilogue: +b2, BatchNorm (eval), relu, residual
#pragma unroll
      for (int v = 0; v < 8; ++v) {
        const int row = v + 8 * hi;
        const int col = nt * 16 + lo;
        const float z  = acc[v] + b2[col];
        const float sc = gamma[col] * rsqrtf(var[col] + 1e-5f);
        const float zz = (z - mean[col]) * sc + beta[col];
        const size_t idx = (size_t)(r0 + row) * HD + col;
        h[idx] = fmaxf(zz, 0.0f) + h[idx];
      }
    }
  }
}

// ---------------------------------------------------------------------------
__global__ __launch_bounds__(256) void pool_zero_kernel(
    float* __restrict__ pooled, float* __restrict__ cnt, int G) {
  const int t = blockIdx.x * blockDim.x + threadIdx.x;
  if (t < G * HD) pooled[t] = 0.0f;
  if (t < G) cnt[t] = 0.0f;
}

__global__ __launch_bounds__(256) void pool_sum_kernel(
    const float* __restrict__ h, const int* __restrict__ batch,
    float* __restrict__ pooled, float* __restrict__ cnt, int N) {
  const int t = blockIdx.x * blockDim.x + threadIdx.x;
  const int node = t >> 5;
  if (node >= N) return;
  const int lane = t & 31;
  const int c = lane * 4;
  const int g = batch[node];
  const float4 hv = *(const float4*)(h + (size_t)node * HD + c);
  float* pp = pooled + (size_t)g * HD + c;
  atomAddF(pp + 0, hv.x); atomAddF(pp + 1, hv.y);
  atomAddF(pp + 2, hv.z); atomAddF(pp + 3, hv.w);
  if (lane == 0) atomAddF(&cnt[g], 1.0f);
}

// out = concat(pooled/cnt, relu(global_x@gp_W+gp_b)) @ gc_W + gc_b
__global__ __launch_bounds__(256) void head_kernel(
    const float* __restrict__ pooled, const float* __restrict__ cnt,
    const float* __restrict__ global_x, const float* __restrict__ gp_W,
    const float* __restrict__ gp_b, const float* __restrict__ gc_W,
    const float* __restrict__ gc_b, float* __restrict__ out, int G, int OUT) {
  const int t = blockIdx.x * blockDim.x + threadIdx.x;
  if (t >= G * OUT) return;
  const int g = t / OUT;
  const int o = t % OUT;
  const float inv = 1.0f / fmaxf(cnt[g], 1.0f);
  float acc = gc_b[o];
#pragma unroll 8
  for (int j = 0; j < HD; ++j)
    acc += pooled[(size_t)g * HD + j] * inv * gc_W[j * OUT + o];
#pragma unroll
  for (int j = 0; j < 32; ++j) {
    float gv = gp_b[j];
#pragma unroll
    for (int k = 0; k < 6; ++k) gv += global_x[g * 6 + k] * gp_W[k * 32 + j];
    acc += fmaxf(gv, 0.0f) * gc_W[(HD + j) * OUT + o];
  }
  out[t] = acc;
}

// ---------------------------------------------------------------------------
extern "C" void kernel_launch(void* const* d_in, const int* in_sizes, int n_in,
                              void* d_out, int out_size, void* d_ws, size_t ws_size,
                              hipStream_t stream) {
  const float* x         = (const float*)d_in[0];
  const float* edge_attr = (const float*)d_in[1];
  const float* global_x  = (const float*)d_in[2];
  const int*   edge_idx  = (const int*)  d_in[3];
  const int*   batch     = (const int*)  d_in[4];
  const float* in_W      = (const float*)d_in[5];
  const float* in_b      = (const float*)d_in[6];
  const float* edge_W    = (const float*)d_in[7];
  const float* edge_b    = (const float*)d_in[8];
  const float* eps       = (const float*)d_in[9];
  const float* mlp_W1    = (const float*)d_in[10];
  const float* mlp_b1    = (const float*)d_in[11];
  const float* mlp_W2    = (const float*)d_in[12];
  const float* mlp_b2    = (const float*)d_in[13];
  const float* bn_gamma  = (const float*)d_in[14];
  const float* bn_beta   = (const float*)d_in[15];
  const float* bn_mean   = (const float*)d_in[16];
  const float* bn_var    = (const float*)d_in[17];
  const float* gp_W      = (const float*)d_in[18];
  const float* gp_b      = (const float*)d_in[19];
  const float* gc_W      = (const float*)d_in[20];
  const float* gc_b      = (const float*)d_in[21];
  float* outp = (float*)d_out;

  const int N = in_sizes[0] / 16;
  const int E = in_sizes[1] / 2;
  const int G = in_sizes[2] / 6;
  const int OUT = HD;

  float* h      = (float*)d_ws;                 // N*HD
  float* agg    = h + (size_t)N * HD;           // N*HD
  float* pooled = agg + (size_t)N * HD;         // G*HD
  float* cntp   = pooled + (size_t)G * HD;      // G

  // input projection
  inproj_kernel<<<(N * 32 + 255) / 256, 256, 0, stream>>>(x, in_W, in_b, h, N);

  const int nTiles = (N + 15) / 16;
  const int mlpBlocks = (nTiles + 7) / 8;
  for (int l = 0; l < LNUM; ++l) {
    zero4_kernel<<<((size_t)N * HD / 4 + 255) / 256, 256, 0, stream>>>(
        (float4*)agg, N * HD / 4);
    msg_kernel<<<((size_t)E * 32 + 255) / 256, 256, 0, stream>>>(
        h, edge_attr, edge_idx, edge_W, edge_b, agg, E);
    gine_mlp_kernel<<<mlpBlocks, 256, 0, stream>>>(
        h, agg,
        mlp_W1 + (size_t)l * HD * HD, mlp_b1 + (size_t)l * HD,
        mlp_W2 + (size_t)l * HD * HD, mlp_b2 + (size_t)l * HD,
        bn_gamma + (size_t)l * HD, bn_beta + (size_t)l * HD,
        bn_mean + (size_t)l * HD, bn_var + (size_t)l * HD,
        eps, l, nTiles);
  }

  pool_zero_kernel<<<(G * HD + 255) / 256, 256, 0, stream>>>(pooled, cntp, G);
  pool_sum_kernel<<<(N * 32 + 255) / 256, 256, 0, stream>>>(h, batch, pooled, cntp, N);
  head_kernel<<<(G * OUT + 255) / 256, 256, 0, stream>>>(
      pooled, cntp, global_x, gp_W, gp_b, gc_W, gc_b, outp, G, OUT);

  (void)n_in; (void)out_size; (void)ws_size;
}
